// MultiHeadSlotAttention_64828236366130
// MI455X (gfx1250) — matrix operand order repaired and checked by reading, not compiled, not run on hardware
//
#include <hip/hip_runtime.h>
#include <hip/hip_bf16.h>

#define B_   32
#define N_   4096
#define D_   256
#define H_   8
#define DH_  64
#define S_   16
#define DI_  512
#define HID_ 1024
#define ITERS_ 3
#define EPS_ 1e-8f
#define SCALE_ 0.0625f   // D^-0.5 = 1/16

typedef _Float16 half8  __attribute__((ext_vector_type(8)));
typedef _Float16 half16 __attribute__((ext_vector_type(16)));
typedef float    float8 __attribute__((ext_vector_type(8)));

// ---- WMMA fragment loaders --------------------------------------------------
// A fragment (16x32 f16), ISA 7.12.2: lanes 0-15 row M=lane, elems 0..7 = K0..7,
// elems 8..15 = K16..23; lanes 16-31: K8..15 / K24..31.
__device__ __forceinline__ half16 frag_ld_a(const _Float16* __restrict__ base, int ld) {
    int lane = threadIdx.x & 31;
    int row  = lane & 15;
    int hi   = lane >> 4;
    const _Float16* p = base + (long)row * ld + hi * 8;
    half8 a = *(const half8*)p;
    half8 b = *(const half8*)(p + 16);
    half16 r;
#pragma unroll
    for (int i = 0; i < 8; ++i) { r[i] = a[i]; r[i + 8] = b[i]; }
    return r;
}

// B fragment (32x16 f16) loaded from B^T stored row-major (N x K).
// Same K order as A: lanes 0-15 elems 0..7 = K0..7, elems 8..15 = K16..23;
// lanes 16-31: K8..15 / K24..31; lane's column = lane&15.
__device__ __forceinline__ half16 frag_ld_b(const _Float16* __restrict__ bT, int ld) {
    int lane = threadIdx.x & 31;
    int col  = lane & 15;
    int koff = (lane >> 4) * 8;
    const _Float16* p = bT + (long)col * ld + koff;
    half8 a = *(const half8*)p;
    half8 b = *(const half8*)(p + 16);
    half16 r;
#pragma unroll
    for (int i = 0; i < 8; ++i) { r[i] = a[i]; r[i + 8] = b[i]; }
    return r;
}

__device__ __forceinline__ float8 wmma16(half16 a, half16 b, float8 c) {
    return __builtin_amdgcn_wmma_f32_16x16x32_f16(false, a, false, b, (short)0, c,
                                                  false, false);
}

// ---- small utility kernels --------------------------------------------------
__global__ void k_slots_init(const float* __restrict__ noise, const float* __restrict__ mu,
                             const float* __restrict__ ls, float* __restrict__ slots) {
    int i = blockIdx.x * 256 + threadIdx.x;   // B*S*D
    int d = i & (D_ - 1);
    slots[i] = mu[d] + expf(ls[d]) * noise[i];
}

__global__ void k_cast_f16(const float* __restrict__ src, _Float16* __restrict__ dst, int n) {
    int i = blockIdx.x * 256 + threadIdx.x;
    if (i < n) dst[i] = (_Float16)src[i];
}

// LayerNorm of x rows (D=256), one wave per row, output f16
__global__ void k_ln_cast(const float* __restrict__ x, const float* __restrict__ g,
                          const float* __restrict__ bb, _Float16* __restrict__ xn) {
    int wave = threadIdx.x >> 5;
    int lane = threadIdx.x & 31;
    long row = (long)blockIdx.x * 8 + wave;
    const float* xr = x + row * D_;
    float v[8], s = 0.f, s2 = 0.f;
#pragma unroll
    for (int i = 0; i < 8; ++i) { v[i] = xr[lane + 32 * i]; s += v[i]; s2 += v[i] * v[i]; }
#pragma unroll
    for (int m = 16; m > 0; m >>= 1) { s += __shfl_xor(s, m, 32); s2 += __shfl_xor(s2, m, 32); }
    float mean = s * (1.0f / D_);
    float var  = s2 * (1.0f / D_) - mean * mean;
    float rstd = rsqrtf(var + 1e-5f);
    _Float16* orow = xn + row * D_;
#pragma unroll
    for (int i = 0; i < 8; ++i) {
        int c = lane + 32 * i;
        orow[c] = (_Float16)((v[i] - mean) * rstd * g[c] + bb[c]);
    }
}

// ---- K / V projection: [32 rows x 64 cols] per wave, K=256 ------------------
// transposed==0: out[row][col]  (k, [B*N, DI] f16)
// transposed==1: out = V^T layout [B,H,DH,N] f16
__global__ void k_proj(const _Float16* __restrict__ xn, const _Float16* __restrict__ Wh,
                       const float* __restrict__ bias, _Float16* __restrict__ out,
                       int transposed) {
    int lane  = threadIdx.x & 31;
    int col64 = threadIdx.x >> 5;            // 0..7 -> 64-col group
    long row0 = (long)blockIdx.x * 32;       // 32 rows per wave-row-tile
    int colbase = col64 * 64;
    float8 acc[2][4];
#pragma unroll
    for (int gI = 0; gI < 4; ++gI) {
        float bv0 = bias[colbase + gI * 16 + (lane & 15)];
#pragma unroll
        for (int mi = 0; mi < 2; ++mi)
#pragma unroll
            for (int r = 0; r < 8; ++r) acc[mi][gI][r] = bv0;
    }
    const _Float16* a0 = xn + row0 * D_;
    const _Float16* a1 = xn + (row0 + 16) * D_;
#pragma unroll 1
    for (int kc = 0; kc < D_ / 32; ++kc) {
        // issue all loads for this k-chunk first, then the WMMA group
        half16 af0 = frag_ld_a(a0 + kc * 32, D_);
        half16 af1 = frag_ld_a(a1 + kc * 32, D_);
        half16 bf0 = frag_ld_b(Wh + (long)(colbase +  0) * D_ + kc * 32, D_);
        half16 bf1 = frag_ld_b(Wh + (long)(colbase + 16) * D_ + kc * 32, D_);
        half16 bf2 = frag_ld_b(Wh + (long)(colbase + 32) * D_ + kc * 32, D_);
        half16 bf3 = frag_ld_b(Wh + (long)(colbase + 48) * D_ + kc * 32, D_);
        acc[0][0] = wmma16(af0, bf0, acc[0][0]);
        acc[0][1] = wmma16(af0, bf1, acc[0][1]);
        acc[0][2] = wmma16(af0, bf2, acc[0][2]);
        acc[0][3] = wmma16(af0, bf3, acc[0][3]);
        acc[1][0] = wmma16(af1, bf0, acc[1][0]);
        acc[1][1] = wmma16(af1, bf1, acc[1][1]);
        acc[1][2] = wmma16(af1, bf2, acc[1][2]);
        acc[1][3] = wmma16(af1, bf3, acc[1][3]);
    }
    int hi = lane >> 4, nn = lane & 15;
    if (!transposed) {
#pragma unroll
        for (int mi = 0; mi < 2; ++mi)
#pragma unroll
            for (int gI = 0; gI < 4; ++gI)
#pragma unroll
                for (int r = 0; r < 8; ++r) {
                    long m = row0 + mi * 16 + r + 8 * hi;
                    out[m * DI_ + colbase + gI * 16 + nn] = (_Float16)acc[mi][gI][r];
                }
    } else {
#pragma unroll
        for (int gI = 0; gI < 4; ++gI) {
            int col = colbase + gI * 16 + nn;
            int h = col >> 6, dh = col & 63;
#pragma unroll
            for (int mi = 0; mi < 2; ++mi)
#pragma unroll
                for (int r = 0; r < 8; ++r) {
                    long grow = row0 + mi * 16 + r + 8 * hi;
                    long b = grow >> 12, n = grow & (N_ - 1);
                    out[(((long)(b * H_ + h) * DH_) + dh) * N_ + n] = (_Float16)acc[mi][gI][r];
                }
        }
    }
}

// ---- q projection with slot LayerNorm (512 rows, tiny) ----------------------
__global__ void k_qproj(const float* __restrict__ slots, const float* __restrict__ g,
                        const float* __restrict__ bvec, const float* __restrict__ Wq,
                        const float* __restrict__ bq, _Float16* __restrict__ q) {
    __shared__ float sn[D_];
    __shared__ float rs[8], rs2[8];
    int row = blockIdx.x, tid = threadIdx.x;
    float v = slots[(long)row * D_ + tid];
    float s = v, s2 = v * v;
#pragma unroll
    for (int m = 16; m > 0; m >>= 1) { s += __shfl_xor(s, m, 32); s2 += __shfl_xor(s2, m, 32); }
    if ((tid & 31) == 0) { rs[tid >> 5] = s; rs2[tid >> 5] = s2; }
    __syncthreads();
    if (tid == 0) {
        float a = 0, c = 0;
        for (int i = 0; i < 8; ++i) { a += rs[i]; c += rs2[i]; }
        rs[0] = a; rs2[0] = c;
    }
    __syncthreads();
    float mean = rs[0] * (1.f / D_);
    float var  = rs2[0] * (1.f / D_) - mean * mean;
    float rstd = rsqrtf(var + 1e-5f);
    sn[tid] = (v - mean) * rstd * g[tid] + bvec[tid];
    __syncthreads();
#pragma unroll
    for (int c = 0; c < 2; ++c) {
        int col = tid + c * 256;
        const float* wr = Wq + (long)col * D_;
        float acc = bq[col];
        for (int j = 0; j < D_; ++j) acc += sn[j] * wr[j];
        q[(long)row * DI_ + col] = (_Float16)acc;
    }
}

// ---- logits = q @ k^T * SCALE : one wave = one (b,h), 32 n-cols -------------
__global__ void k_logits(const _Float16* __restrict__ q, const _Float16* __restrict__ k,
                         float* __restrict__ logits) {
    int wave = threadIdx.x >> 5;
    int lane = threadIdx.x & 31;
    int bh = blockIdx.x >> 4;                    // 0..255 ; 16 blocks per bh
    int nt32 = (blockIdx.x & 15) * 8 + wave;     // 0..127 (32-wide n tile)
    int b = bh >> 3, h = bh & 7;
    const _Float16* qb = q + (long)b * S_ * DI_ + h * DH_;
    const _Float16* kb0 = k + ((long)b * N_ + nt32 * 32) * DI_ + h * DH_;
    const _Float16* kb1 = kb0 + 16 * DI_;
    // q fragments loaded once, reused for both 16-col tiles
    half16 af0 = frag_ld_a(qb, DI_);
    half16 af1 = frag_ld_a(qb + 32, DI_);
    half16 b00 = frag_ld_b(kb0, DI_);
    half16 b01 = frag_ld_b(kb0 + 32, DI_);
    half16 b10 = frag_ld_b(kb1, DI_);
    half16 b11 = frag_ld_b(kb1 + 32, DI_);
    float8 acc0, acc1;
#pragma unroll
    for (int r = 0; r < 8; ++r) { acc0[r] = 0.f; acc1[r] = 0.f; }
    acc0 = wmma16(af0, b00, acc0);
    acc0 = wmma16(af1, b01, acc0);
    acc1 = wmma16(af0, b10, acc1);
    acc1 = wmma16(af1, b11, acc1);
    int hi = lane >> 4, nn = lane & 15;
#pragma unroll
    for (int r = 0; r < 8; ++r) {
        int s = r + 8 * hi;
        long rowbase = ((long)bh * S_ + s) * N_ + nt32 * 32;
        logits[rowbase + nn]      = acc0[r] * SCALE_;
        logits[rowbase + 16 + nn] = acc1[r] * SCALE_;
    }
}

// ---- softmax + eps renorm, one block per (b,h,s) row, attn stored f16 -------
__global__ void k_softmax(const float* __restrict__ logits, _Float16* __restrict__ attn) {
    __shared__ float rbuf[8];
    int row = blockIdx.x, tid = threadIdx.x;
    const float* lr = logits + (long)row * N_;
    float vals[16], m = -1e30f;
#pragma unroll
    for (int i = 0; i < 16; ++i) { vals[i] = lr[tid + 256 * i]; m = fmaxf(m, vals[i]); }
#pragma unroll
    for (int s = 16; s > 0; s >>= 1) m = fmaxf(m, __shfl_xor(m, s, 32));
    if ((tid & 31) == 0) rbuf[tid >> 5] = m;
    __syncthreads();
    if (tid == 0) { float mm = rbuf[0]; for (int i = 1; i < 8; ++i) mm = fmaxf(mm, rbuf[i]); rbuf[0] = mm; }
    __syncthreads();
    m = rbuf[0];
    __syncthreads();
    float s = 0.f;
#pragma unroll
    for (int i = 0; i < 16; ++i) { vals[i] = expf(vals[i] - m); s += vals[i]; }
#pragma unroll
    for (int sh = 16; sh > 0; sh >>= 1) s += __shfl_xor(s, sh, 32);
    if ((tid & 31) == 0) rbuf[tid >> 5] = s;
    __syncthreads();
    if (tid == 0) { float ss = 0; for (int i = 0; i < 8; ++i) ss += rbuf[i]; rbuf[0] = ss; }
    __syncthreads();
    float inv = 1.0f / rbuf[0];
    const float renorm = 1.0f / (1.0f + (float)N_ * EPS_);
    _Float16* ar = attn + (long)row * N_;
#pragma unroll
    for (int i = 0; i < 16; ++i)
        ar[tid + 256 * i] = (_Float16)((vals[i] * inv + EPS_) * renorm);
}

// ---- upd = attn @ v : block = one (b,h); 8 waves split K=4096, LDS reduce ---
__global__ void k_upd(const _Float16* __restrict__ attn, const _Float16* __restrict__ vT,
                      float* __restrict__ upd) {
    __shared__ float part[8][1024];
    int wave = threadIdx.x >> 5;
    int lane = threadIdx.x & 31;
    int bh = blockIdx.x;           // 0..255
    int b = bh >> 3, h = bh & 7;
    float8 acc[4];
#pragma unroll
    for (int gI = 0; gI < 4; ++gI)
#pragma unroll
        for (int r = 0; r < 8; ++r) acc[gI][r] = 0.f;
    const _Float16* ab = attn + (long)bh * S_ * N_;
    const _Float16* vb = vT + (long)bh * DH_ * N_;
#pragma unroll 1
    for (int kc = 0; kc < 16; ++kc) {
        int n0 = wave * 512 + kc * 32;
        half16 af  = frag_ld_a(ab + n0, N_);
        half16 bf0 = frag_ld_b(vb + (long)0  * N_ + n0, N_);
        half16 bf1 = frag_ld_b(vb + (long)16 * N_ + n0, N_);
        half16 bf2 = frag_ld_b(vb + (long)32 * N_ + n0, N_);
        half16 bf3 = frag_ld_b(vb + (long)48 * N_ + n0, N_);
        acc[0] = wmma16(af, bf0, acc[0]);
        acc[1] = wmma16(af, bf1, acc[1]);
        acc[2] = wmma16(af, bf2, acc[2]);
        acc[3] = wmma16(af, bf3, acc[3]);
    }
    int hi = lane >> 4, nn = lane & 15;
#pragma unroll
    for (int gI = 0; gI < 4; ++gI)
#pragma unroll
        for (int r = 0; r < 8; ++r) {
            int s = r + 8 * hi;
            part[wave][s * 64 + gI * 16 + nn] = acc[gI][r];
        }
    __syncthreads();
    for (int e = threadIdx.x; e < 1024; e += 256) {
        float t = 0.f;
#pragma unroll
        for (int w = 0; w < 8; ++w) t += part[w][e];
        int s = e >> 6, dh = e & 63;
        upd[(long)(b * S_ + s) * DI_ + h * DH_ + dh] = t;
    }
}

// ---- small slot-update chain (512 rows) -------------------------------------
__global__ void k_wc(const float* __restrict__ upd, const float* __restrict__ Wc,
                     const float* __restrict__ bc, float* __restrict__ updc) {
    __shared__ float u[DI_];
    int row = blockIdx.x, tid = threadIdx.x;
    u[tid]       = upd[(long)row * DI_ + tid];
    u[tid + 256] = upd[(long)row * DI_ + tid + 256];
    __syncthreads();
    const float* wr = Wc + (long)tid * DI_;
    float acc = bc[tid];
    for (int j = 0; j < DI_; ++j) acc += u[j] * wr[j];
    updc[(long)row * D_ + tid] = acc;
}

__global__ void k_gru(const float* __restrict__ updc, float* __restrict__ slots,
                      const float* __restrict__ wih, const float* __restrict__ whh,
                      const float* __restrict__ bih, const float* __restrict__ bhh) {
    __shared__ float u[D_], hp[D_];
    int row = blockIdx.x, d = threadIdx.x;
    u[d]  = updc[(long)row * D_ + d];
    hp[d] = slots[(long)row * D_ + d];
    __syncthreads();
    float ir = bih[d], iz = bih[D_ + d], in_ = bih[2 * D_ + d];
    float hr = bhh[d], hz = bhh[D_ + d], hn = bhh[2 * D_ + d];
    const float* w0 = wih + (long)d * D_;
    const float* w1 = wih + (long)(D_ + d) * D_;
    const float* w2 = wih + (long)(2 * D_ + d) * D_;
    const float* v0 = whh + (long)d * D_;
    const float* v1 = whh + (long)(D_ + d) * D_;
    const float* v2 = whh + (long)(2 * D_ + d) * D_;
    for (int j = 0; j < D_; ++j) {
        float uj = u[j], hj = hp[j];
        ir += uj * w0[j]; iz += uj * w1[j]; in_ += uj * w2[j];
        hr += hj * v0[j]; hz += hj * v1[j]; hn += hj * v2[j];
    }
    float r = 1.f / (1.f + expf(-(ir + hr)));
    float z = 1.f / (1.f + expf(-(iz + hz)));
    float n = tanhf(in_ + r * hn);
    slots[(long)row * D_ + d] = (1.f - z) * n + z * hp[d];
}

__global__ void k_ff1(const float* __restrict__ slots, const float* __restrict__ g,
                      const float* __restrict__ bvec, const float* __restrict__ W1,
                      const float* __restrict__ b1, float* __restrict__ h1) {
    __shared__ float sn[D_];
    __shared__ float rs[8], rs2[8];
    int row = blockIdx.x, tid = threadIdx.x;
    float v = slots[(long)row * D_ + tid];
    float s = v, s2 = v * v;
#pragma unroll
    for (int m = 16; m > 0; m >>= 1) { s += __shfl_xor(s, m, 32); s2 += __shfl_xor(s2, m, 32); }
    if ((tid & 31) == 0) { rs[tid >> 5] = s; rs2[tid >> 5] = s2; }
    __syncthreads();
    if (tid == 0) {
        float a = 0, c = 0;
        for (int i = 0; i < 8; ++i) { a += rs[i]; c += rs2[i]; }
        rs[0] = a; rs2[0] = c;
    }
    __syncthreads();
    float mean = rs[0] * (1.f / D_);
    float var  = rs2[0] * (1.f / D_) - mean * mean;
    float rstd = rsqrtf(var + 1e-5f);
    sn[tid] = (v - mean) * rstd * g[tid] + bvec[tid];
    __syncthreads();
#pragma unroll
    for (int c = 0; c < 4; ++c) {
        int col = tid + 256 * c;
        const float* wr = W1 + (long)col * D_;
        float acc = b1[col];
        for (int j = 0; j < D_; ++j) acc += sn[j] * wr[j];
        h1[(long)row * HID_ + col] = fmaxf(acc, 0.f);
    }
}

__global__ void k_ff2(const float* __restrict__ h1, const float* __restrict__ W2,
                      const float* __restrict__ b2, float* __restrict__ slots) {
    __shared__ float hh[HID_];
    int row = blockIdx.x, tid = threadIdx.x;
#pragma unroll
    for (int c = 0; c < 4; ++c) hh[tid + 256 * c] = h1[(long)row * HID_ + tid + 256 * c];
    __syncthreads();
    const float* wr = W2 + (long)tid * HID_;
    float acc = b2[tid];
    for (int j = 0; j < HID_; ++j) acc += hh[j] * wr[j];
    slots[(long)row * D_ + tid] += acc;
}

// ---- launch -----------------------------------------------------------------
extern "C" void kernel_launch(void* const* d_in, const int* in_sizes, int n_in,
                              void* d_out, int out_size, void* d_ws, size_t ws_size,
                              hipStream_t stream) {
    (void)in_sizes; (void)n_in; (void)out_size; (void)ws_size;
    const float* x      = (const float*)d_in[0];
    const float* noise  = (const float*)d_in[1];
    const float* mu     = (const float*)d_in[2];
    const float* ls     = (const float*)d_in[3];
    const float* ln_in_g = (const float*)d_in[4];
    const float* ln_in_b = (const float*)d_in[5];
    const float* ln_s_g  = (const float*)d_in[6];
    const float* ln_s_b  = (const float*)d_in[7];
    const float* ln_ff_g = (const float*)d_in[8];
    const float* ln_ff_b = (const float*)d_in[9];
    const float* Wq = (const float*)d_in[10];
    const float* bq = (const float*)d_in[11];
    const float* Wk = (const float*)d_in[12];
    const float* bk = (const float*)d_in[13];
    const float* Wv = (const float*)d_in[14];
    const float* bv = (const float*)d_in[15];
    const float* Wc = (const float*)d_in[16];
    const float* bc = (const float*)d_in[17];
    const float* wih = (const float*)d_in[18];
    const float* whh = (const float*)d_in[19];
    const float* bih = (const float*)d_in[20];
    const float* bhh = (const float*)d_in[21];
    const float* W1 = (const float*)d_in[22];
    const float* b1 = (const float*)d_in[23];
    const float* W2 = (const float*)d_in[24];
    const float* b2 = (const float*)d_in[25];

    char* p = (char*)d_ws;
    auto carve = [&](size_t bytes) -> char* {
        char* r = p; p += (bytes + 255) & ~(size_t)255; return r;
    };
    _Float16* xn    = (_Float16*)carve((size_t)B_ * N_ * D_ * 2);
    _Float16* kf    = (_Float16*)carve((size_t)B_ * N_ * DI_ * 2);
    _Float16* vT    = (_Float16*)carve((size_t)B_ * N_ * DI_ * 2);
    _Float16* wk16  = (_Float16*)carve((size_t)DI_ * D_ * 2);
    _Float16* wv16  = (_Float16*)carve((size_t)DI_ * D_ * 2);
    float*    slots = (float*)carve((size_t)B_ * S_ * D_ * 4);
    _Float16* qf    = (_Float16*)carve((size_t)B_ * S_ * DI_ * 2);
    float*    logits = (float*)carve((size_t)B_ * H_ * S_ * N_ * 4);
    _Float16* attn  = (_Float16*)carve((size_t)B_ * H_ * S_ * N_ * 2);
    float*    upd   = (float*)carve((size_t)B_ * S_ * DI_ * 4);
    float*    updc  = (float*)carve((size_t)B_ * S_ * D_ * 4);
    float*    h1    = (float*)carve((size_t)B_ * S_ * HID_ * 4);

    k_slots_init<<<(B_ * S_ * D_) / 256, 256, 0, stream>>>(noise, mu, ls, slots);
    k_cast_f16<<<(DI_ * D_) / 256, 256, 0, stream>>>(Wk, wk16, DI_ * D_);
    k_cast_f16<<<(DI_ * D_) / 256, 256, 0, stream>>>(Wv, wv16, DI_ * D_);
    k_ln_cast<<<(B_ * N_) / 8, 256, 0, stream>>>(x, ln_in_g, ln_in_b, xn);
    k_proj<<<(B_ * N_) / 32, 256, 0, stream>>>(xn, wk16, bk, kf, 0);
    k_proj<<<(B_ * N_) / 32, 256, 0, stream>>>(xn, wv16, bv, vT, 1);

    for (int it = 0; it < ITERS_; ++it) {
        k_qproj<<<B_ * S_, 256, 0, stream>>>(slots, ln_s_g, ln_s_b, Wq, bq, qf);
        k_logits<<<(B_ * H_ * (N_ / 32)) / 8, 256, 0, stream>>>(qf, kf, logits);
        k_softmax<<<B_ * H_ * S_, 256, 0, stream>>>(logits, attn);
        k_upd<<<B_ * H_, 256, 0, stream>>>(attn, vT, upd);
        k_wc<<<B_ * S_, 256, 0, stream>>>(upd, Wc, bc, updc);
        k_gru<<<B_ * S_, 256, 0, stream>>>(updc, slots, wih, whh, bih, bhh);
        k_ff1<<<B_ * S_, 256, 0, stream>>>(slots, ln_ff_g, ln_ff_b, W1, b1, h1);
        k_ff2<<<B_ * S_, 256, 0, stream>>>(h1, W2, b2, slots);
    }
    hipMemcpyAsync(d_out, slots, (size_t)B_ * S_ * D_ * 4,
                   hipMemcpyDeviceToDevice, stream);
}
